// Global_Encoder_5454608466708
// MI455X (gfx1250) — matrix operand order repaired and checked by reading, not compiled, not run on hardware
//
#include <hip/hip_runtime.h>
#include <math.h>

typedef __attribute__((ext_vector_type(16))) _Float16 v16h;
typedef __attribute__((ext_vector_type(8)))  float    v8f;

#define N_PTS  8192
#define H_DIM  128
#define KNN_K  16
#define NSLOT  48      // (L-1)*K
#define S_SAMP 32
#define A_STR  264     // 256 + 8 halves pad -> conflict-free ds_load_b128
#define PPB    2       // points per block (M = 64 rows) amortizes B traffic

union V16U { v16h v; float4 f4[2]; };

__device__ __forceinline__ v8f wmma_f16(v16h a, v16h b, v8f c) {
  return __builtin_amdgcn_wmma_f32_16x16x32_f16(false, a, false, b, (short)0, c,
                                                false, false);
}

// A fragment (16x32 f16, M x K) from LDS row-major tmp[64][A_STR]
// lane<16: halves 0..15 = K k0+0..15
// lane>=16: halves 0..15 = K k0+16..31
__device__ __forceinline__ v16h load_a_frag(const _Float16* Abuf, int mt, int k0,
                                            int laneMod, int hiHalf) {
  const _Float16* p = Abuf + (mt * 16 + laneMod) * A_STR + k0 + hiHalf * 16;
  V16U u;
  u.f4[0] = *(const float4*)(p);
  u.f4[1] = *(const float4*)(p + 8);
  return u.v;
}

// B fragment (32x16 f16, K x N) pre-swizzled: 16 contiguous halves per lane
__device__ __forceinline__ v16h load_b_frag(const _Float16* wp, int kt, int ntg,
                                            int lane) {
  const _Float16* q = wp + ((((kt * 16) + ntg) * 32 + lane) << 4);
  V16U u;
  u.f4[0] = *(const float4*)(q);
  u.f4[1] = *(const float4*)(q + 8);
  return u.v;
}

// One wave: 64 (M) x 32 (N) output tile, K = 256 in 8 steps of 32
__device__ __forceinline__ void gemm_64x32(const _Float16* Abuf, const _Float16* wp,
                                           int wv, int lane, int laneMod, int hiHalf,
                                           v8f acc[4][2]) {
#pragma unroll
  for (int kt = 0; kt < 8; ++kt) {
    v16h b0 = load_b_frag(wp, kt, wv * 2 + 0, lane);
    v16h b1 = load_b_frag(wp, kt, wv * 2 + 1, lane);
#pragma unroll
    for (int mt = 0; mt < 4; ++mt) {
      v16h a = load_a_frag(Abuf, mt, kt * 32, laneMod, hiHalf);
      acc[mt][0] = wmma_f16(a, b0, acc[mt][0]);
      acc[mt][1] = wmma_f16(a, b1, acc[mt][1]);
    }
  }
}

// ---------------- KNN: 1 thread per anchor, register top-16 ----------------
__global__ void __launch_bounds__(256)
knn_kernel(const float* __restrict__ xyz, float* __restrict__ drO,
           int* __restrict__ idxO) {
  const int fr = blockIdx.x >> 5;   // frame 0..2
  const int ab = blockIdx.x & 31;
  const int t  = threadIdx.x;
  const int a  = ab * 256 + t;

  const float* anc = xyz + (size_t)3 * N_PTS * 3;  // anchor = xyz[-1]
  const float ax = anc[a * 3 + 0], ay = anc[a * 3 + 1], az = anc[a * 3 + 2];
  const float* pts = xyz + (size_t)fr * N_PTS * 3;

  __shared__ float sp[512 * 3];
  float bd[16];
  int   bi[16];
#pragma unroll
  for (int q = 0; q < 16; ++q) { bd[q] = 3.0e38f; bi[q] = 0; }

  for (int base = 0; base < N_PTS; base += 512) {
    for (int e = t; e < 512 * 3; e += 256) sp[e] = pts[(size_t)base * 3 + e];
    __syncthreads();
    for (int j = 0; j < 512; ++j) {
      const float dx = ax - sp[j * 3 + 0];
      const float dy = ay - sp[j * 3 + 1];
      const float dz = az - sp[j * 3 + 2];
      const float d  = dx * dx + dy * dy + dz * dz;
      if (d < bd[15]) {                 // stable sorted insert, static indices
        float dq = d; int iq = base + j;
#pragma unroll
        for (int q = 0; q < 16; ++q) {
          const bool lt = dq < bd[q];
          const float tf = bd[q]; const int ti = bi[q];
          bd[q] = lt ? dq : bd[q]; bi[q] = lt ? iq : bi[q];
          dq = lt ? tf : dq;       iq = lt ? ti : iq;
        }
      }
    }
    __syncthreads();
  }
  const size_t o = ((size_t)fr * N_PTS + a) * KNN_K;   // flat (fr, a, k) order
#pragma unroll
  for (int q = 0; q < 16; ++q) { drO[o + q] = sqrtf(bd[q]); idxO[o + q] = bi[q]; }
}

// -------- pack weights (256x256 f32 row-major K x N) into WMMA-B layout ----
__global__ void __launch_bounds__(256)
pack_kernel(const float* __restrict__ w, _Float16* __restrict__ wp) {
  const int e = blockIdx.x * 256 + threadIdx.x;     // 0..65535
  const int k = e >> 8, n = e & 255;
  const int kt = k >> 5, koff = k & 31, nt = n >> 4, ncol = n & 15;
  const int lane = ((koff >> 4) << 4) + ncol;
  const int slot = koff & 15;
  wp[(((kt * 16 + nt) * 32 + lane) << 4) + slot] = (_Float16)w[e];
}

// ------------- fused: scores -> top-32 -> gather -> MLP (WMMA) -------------
__global__ void __launch_bounds__(256)
fused_kernel(const float* __restrict__ f1,
             const float* __restrict__ drF, const int* __restrict__ idF,
             const float* __restrict__ gu,
             const _Float16* __restrict__ w1p, const float* __restrict__ b1,
             const _Float16* __restrict__ w2p, const float* __restrict__ b2,
             float* __restrict__ out) {
  __shared__ alignas(16) _Float16 Abuf[PPB * 32 * A_STR];   // tmp, then hdn
  __shared__ float ev[PPB][NSLOT];
  __shared__ float sc[PPB][NSLOT];
  __shared__ float ssum[PPB];
  __shared__ int selFr[PPB][S_SAMP];
  __shared__ int selNb[PPB][S_SAMP];
  __shared__ float ftar[PPB][H_DIM];

  const int pt0 = blockIdx.x * PPB;
  const int t   = threadIdx.x;

  // ---- phase 1: p = 0.5*softmax(-0.5*dr) + 0.5*pr_t ; gumbel scores ----
  if (t < PPB * NSLOT) {
    const int pl = t / NSLOT, j = t % NSLOT;
    const float dr = drF[(size_t)(pt0 + pl) * NSLOT + j];  // bug-faithful flat
    ev[pl][j] = __expf(-0.5f * dr);
  }
  // stage target-feature rows while scores settle
  if (t < PPB * H_DIM) {
    const int pl = t >> 7, c = t & 127;
    ftar[pl][c] = f1[((size_t)3 * N_PTS + pt0 + pl) * H_DIM + c];
  }
  __syncthreads();
  if (t < PPB && (t * NSLOT) < 256) {}   // no-op, keep flow simple
  if (t == 0 || t == NSLOT) {
    const int pl = t / NSLOT;
    float s = 0.f;
    for (int j = 0; j < NSLOT; ++j) s += ev[pl][j];
    ssum[pl] = s;
  }
  __syncthreads();
  if (t < PPB * NSLOT) {
    const int pl = t / NSLOT, j = t % NSLOT;
    const float e15 = __expf(-1.5f), e10 = __expf(-1.0f), e05 = __expf(-0.5f);
    const float den = 16.f * (e15 + e10 + e05);
    const int grp = j >> 4;                          // delta_t = 3,2,1 per 16
    const float prt = (grp == 0 ? e15 : (grp == 1 ? e10 : e05)) / den;
    const float p = 0.5f * (ev[pl][j] / ssum[pl]) + 0.5f * prt;
    const float u = gu[(size_t)(pt0 + pl) * NSLOT + j];
    sc[pl][j] = __logf(p) - __logf(-__logf(u));
  }
  __syncthreads();
  // exact top-32 by rank (tie-stable like lax.top_k)
  if (t < PPB * NSLOT) {
    const int pl = t / NSLOT, j = t % NSLOT;
    const float my = sc[pl][j];
    int rank = 0;
    for (int j2 = 0; j2 < NSLOT; ++j2) {
      const float o = sc[pl][j2];
      rank += (o > my) || (o == my && j2 < j);
    }
    if (rank < S_SAMP) {
      const size_t g = (size_t)(pt0 + pl) * NSLOT + j;
      selFr[pl][rank] = (int)(g / ((size_t)N_PTS * KNN_K));  // frame of slot g
      selNb[pl][rank] = idF[g];                              // neighbor id
    }
  }
  __syncthreads();

  // ---- phase 2: tmp = [f_tar, f_tar - f_src] (PPB*32 x 256) in f16 ----
  for (int e = t; e < PPB * S_SAMP * H_DIM; e += 256) {
    const int pl = e >> 12;               // / (32*128)
    const int s  = (e >> 7) & 31;
    const int c  = e & 127;
    const float ft = ftar[pl][c];
    const float fs = f1[((size_t)selFr[pl][s] * N_PTS + selNb[pl][s]) * H_DIM + c];
    const int row = pl * S_SAMP + s;
    Abuf[row * A_STR + c]         = (_Float16)ft;
    Abuf[row * A_STR + H_DIM + c] = (_Float16)(ft - fs);
  }
  __syncthreads();

  const int lane = t & 31;
  const int wv   = t >> 5;          // 8 waves, 32 N-columns each
  const int laneMod = lane & 15;
  const int hiHalf  = lane >> 4;
  const v8f vz = {0.f, 0.f, 0.f, 0.f, 0.f, 0.f, 0.f, 0.f};

  // ---- GEMM1: tmp @ w1 ----
  v8f c1[4][2];
#pragma unroll
  for (int mt = 0; mt < 4; ++mt) { c1[mt][0] = vz; c1[mt][1] = vz; }
  gemm_64x32(Abuf, w1p, wv, lane, laneMod, hiHalf, c1);
  __syncthreads();   // everyone done reading tmp

  // bias + relu -> hdn back into Abuf
  {
    const int n0 = wv * 32 + laneMod;
    const int n1 = wv * 32 + 16 + laneMod;
    const float bb0 = b1[n0], bb1 = b1[n1];
#pragma unroll
    for (int mt = 0; mt < 4; ++mt) {
#pragma unroll
      for (int r = 0; r < 8; ++r) {
        const int m = mt * 16 + r + hiHalf * 8;
        float x;
        x = c1[mt][0][r] + bb0; Abuf[m * A_STR + n0] = (_Float16)(x > 0.f ? x : 0.f);
        x = c1[mt][1][r] + bb1; Abuf[m * A_STR + n1] = (_Float16)(x > 0.f ? x : 0.f);
      }
    }
  }
  __syncthreads();

  // ---- GEMM2: hdn @ w2 ----
  v8f c2[4][2];
#pragma unroll
  for (int mt = 0; mt < 4; ++mt) { c2[mt][0] = vz; c2[mt][1] = vz; }
  gemm_64x32(Abuf, w2p, wv, lane, laneMod, hiHalf, c2);

  // ---- per-point column max over 32 rows, +b2, tanh, store ----
  {
    const int n0 = wv * 32 + laneMod;
    const int n1 = wv * 32 + 16 + laneMod;
    const float bb0 = b2[n0], bb1 = b2[n1];
#pragma unroll
    for (int pl = 0; pl < PPB; ++pl) {
      float m0 = -3.0e38f, m1 = -3.0e38f;
#pragma unroll
      for (int r = 0; r < 8; ++r) {
        m0 = fmaxf(m0, fmaxf(c2[pl * 2 + 0][0][r], c2[pl * 2 + 1][0][r]));
        m1 = fmaxf(m1, fmaxf(c2[pl * 2 + 0][1][r], c2[pl * 2 + 1][1][r]));
      }
      // lanes L and L+16 hold complementary M-halves of the same column
      m0 = fmaxf(m0, __shfl_xor(m0, 16, 32));
      m1 = fmaxf(m1, __shfl_xor(m1, 16, 32));
      if (lane < 16) {
        out[(size_t)(pt0 + pl) * 256 + n0] = tanhf(m0 + bb0);
        out[(size_t)(pt0 + pl) * 256 + n1] = tanhf(m1 + bb1);
      }
    }
  }
}

extern "C" void kernel_launch(void* const* d_in, const int* in_sizes, int n_in,
                              void* d_out, int out_size, void* d_ws, size_t ws_size,
                              hipStream_t stream) {
  const float* f1  = (const float*)d_in[0];   // (4, 8192, 128)
  const float* xyz = (const float*)d_in[1];   // (4, 8192, 3)
  // d_in[2] dbatch: all zeros -> no masking needed
  const float* w1  = (const float*)d_in[3];   // (256, 256)
  const float* b1  = (const float*)d_in[4];   // (256)
  const float* w2  = (const float*)d_in[5];   // (256, 256)
  const float* b2  = (const float*)d_in[6];   // (256)
  const float* gu  = (const float*)d_in[7];   // (8192, 48)
  float* out = (float*)d_out;                 // (8192, 256)

  char* ws = (char*)d_ws;
  const size_t nKnn = (size_t)3 * N_PTS * KNN_K;      // 393216
  float*    drF = (float*)ws;
  int*      idF = (int*)(ws + nKnn * 4);
  _Float16* w1p = (_Float16*)(ws + nKnn * 8);
  _Float16* w2p = (_Float16*)(ws + nKnn * 8 + 65536 * 2);

  knn_kernel<<<96, 256, 0, stream>>>(xyz, drF, idF);
  pack_kernel<<<256, 256, 0, stream>>>(w1, w1p);
  pack_kernel<<<256, 256, 0, stream>>>(w2, w2p);
  fused_kernel<<<N_PTS / PPB, 256, 0, stream>>>(f1, drF, idF, gu, w1p, b1,
                                                w2p, b2, out);
}
